// Eyettention_readerID_85014582657184
// MI455X (gfx1250) — matrix operand order repaired and checked by reading, not compiled, not run on hardware
//
#include <hip/hip_runtime.h>

// ---------------------------------------------------------------------------
// Eyettention forward for MI455X (gfx1250).
// All GEMMs -> V_WMMA_F32_16X16X32_BF16 with fragment-direct b128 loads:
// weights pre-packed transposed+padded; A fragments are 2x b128, B built from 16 strided 16-bit loads.
// No LDS staging: weights (<10MB bf16) + activation tiles live in the 192MB L2.
// ---------------------------------------------------------------------------

typedef __bf16 bf16_t;
typedef __attribute__((ext_vector_type(16))) __bf16 v16bf;
typedef __attribute__((ext_vector_type(8)))  __bf16 v8bf;
typedef __attribute__((ext_vector_type(8)))  float  v8f;

#define BATCH 512
#define SN    24
#define SPL   40
#define TDEC  39
#define EMB   768
#define HID   128
#define H2    64
#define OUTC  71    // 45 loc + 24 attn + 1 dur + 1 land
#define DEP   352   // padded row stride of 'de' (302 -> 352; de+45 view needs 45+288)
#define DINP  832   // padded row stride of decoder input (802 -> 832)

__device__ __forceinline__ bf16_t f2b(float f) { return (bf16_t)f; }
__device__ __forceinline__ float  b2f(bf16_t b) { return (float)b; }
__device__ __forceinline__ float  sigm(float x) { return 1.f / (1.f + __expf(-x)); }

// ---------------------------------------------------------------------------
// GEMM: C[M,N] = act( A[M,*](bf16, lda>=Kp, zero-padded K tail)
//                     x Wt[Kp,Np](bf16, pre-transposed W^T, zero-padded) + init )
// init = bias[N] (flags&2==0) or outF accumulate (flags&2).  act = relu (flags&1).
// Block: 256 threads / 8 waves -> 64(M) x 128(N) C macro-tile.
// Wave: 2x2 16x16 sub-tiles -> 4 WMMA per 32-K slab, 4 b128 A loads + 32 strided 16-bit B loads.
// Requires M % 64 == 0 (true for all call sites: 512, 12288, 19968).
// ---------------------------------------------------------------------------
__global__ __launch_bounds__(256) void gemm_bf16_wmma_kernel(
    const bf16_t* __restrict__ A, int lda,
    const bf16_t* __restrict__ Wt, int ldwt,
    const float* __restrict__ bias,
    int M, int N, int Kp, int flags,
    float* __restrict__ outF, int ldf,
    bf16_t* __restrict__ outB, int ldb)
{
    const int tid  = threadIdx.x;
    const int wave = tid >> 5;
    const int lane = tid & 31;
    const int hi   = lane >> 4;
    const int l15  = lane & 15;
    const int m0   = blockIdx.y * 64 + (wave >> 2) * 32;   // wave M base (2 sub-tiles)
    const int n0   = blockIdx.x * 128 + (wave & 3) * 32;   // wave N base (2 sub-tiles)

    // ---- init accumulators (C layout: VGPR r -> m=r+8*hi, n=lane&15) ----
    v8f acc[2][2];
#pragma unroll
    for (int ms = 0; ms < 2; ++ms)
#pragma unroll
        for (int ns = 0; ns < 2; ++ns)
#pragma unroll
            for (int r = 0; r < 8; ++r) {
                const int gm = m0 + ms * 16 + r + 8 * hi;
                const int gn = n0 + ns * 16 + l15;
                float v = 0.f;
                if (bias && gn < N) v = bias[gn];
                if ((flags & 2) && gn < N) v = outF[(long)gm * ldf + gn];
                acc[ms][ns][r] = v;
            }

    // A fragment rows: lane m = base + (lane&15); two contiguous 8-elem runs
    const bf16_t* pa0 = A + (long)(m0 + l15) * lda + 8 * hi;
    const bf16_t* pa1 = pa0 + (long)16 * lda;
    // B fragment column: lane n = n0 + (lane&15); k = kb + 8*hi + {0..7, 16..23} (strided rows)
    const bf16_t* pw = Wt + (long)(8 * hi) * ldwt + n0 + l15;

    for (int kb = 0; kb < Kp; kb += 32) {
        const v8bf a0lo = *(const v8bf*)(pa0 + kb);
        const v8bf a0hi = *(const v8bf*)(pa0 + kb + 16);
        const v8bf a1lo = *(const v8bf*)(pa1 + kb);
        const v8bf a1hi = *(const v8bf*)(pa1 + kb + 16);
        const bf16_t* pwk = pw + (long)kb * ldwt;
        v16bf fb0, fb1;
#pragma unroll
        for (int s = 0; s < 16; ++s) {
            const long ko = (long)((s & 7) + 16 * (s >> 3)) * ldwt;
            fb0[s] = pwk[ko];
            fb1[s] = pwk[ko + 16];
        }
        if (kb + 32 < Kp) {   // pull next A slab toward L0/L2 (global_prefetch_b8)
            __builtin_prefetch((const void*)(pa0 + kb + 32), 0, 3);
            __builtin_prefetch((const void*)(pa1 + kb + 32), 0, 3);
        }
        const v16bf fa0 = __builtin_shufflevector(a0lo, a0hi, 0,1,2,3,4,5,6,7,8,9,10,11,12,13,14,15);
        const v16bf fa1 = __builtin_shufflevector(a1lo, a1hi, 0,1,2,3,4,5,6,7,8,9,10,11,12,13,14,15);

        acc[0][0] = __builtin_amdgcn_wmma_f32_16x16x32_bf16(false, fa0, false, fb0, (short)0, acc[0][0], false, false);
        acc[0][1] = __builtin_amdgcn_wmma_f32_16x16x32_bf16(false, fa0, false, fb1, (short)0, acc[0][1], false, false);
        acc[1][0] = __builtin_amdgcn_wmma_f32_16x16x32_bf16(false, fa1, false, fb0, (short)0, acc[1][0], false, false);
        acc[1][1] = __builtin_amdgcn_wmma_f32_16x16x32_bf16(false, fa1, false, fb1, (short)0, acc[1][1], false, false);
    }

    // ---- epilogue ----
#pragma unroll
    for (int ms = 0; ms < 2; ++ms)
#pragma unroll
        for (int ns = 0; ns < 2; ++ns)
#pragma unroll
            for (int r = 0; r < 8; ++r) {
                float v = acc[ms][ns][r];
                if (flags & 1) v = fmaxf(v, 0.f);
                const int gm = m0 + ms * 16 + r + 8 * hi;
                const int gn = n0 + ns * 16 + l15;
                if (gn < N) {
                    if (outF) outF[(long)gm * ldf + gn] = v;
                    if (outB) outB[(long)gm * ldb + gn] = f2b(v);
                }
            }
}

// ---------------------------------------------------------------------------
// Elementwise / glue kernels
// ---------------------------------------------------------------------------
__global__ void zero_u32_kernel(unsigned int* p, long n) {
    long i = (long)blockIdx.x * blockDim.x + threadIdx.x;
    if (i < n) p[i] = 0u;
}
__global__ void add_f32_kernel(const float* a, const float* b, float* o, int n) {
    int i = blockIdx.x * blockDim.x + threadIdx.x;
    if (i < n) o[i] = a[i] + b[i];
}
// pack weight W[N][K] (f32) -> Wt[Kp][Np] (bf16, transposed, zero-padded)
__global__ void pack_wt_kernel(const float* __restrict__ W, bf16_t* __restrict__ Wt,
                               int N, int K, int Kp, int Np) {
    long i = (long)blockIdx.x * blockDim.x + threadIdx.x;
    if (i >= (long)Kp * Np) return;
    const int n = (int)(i % Np);
    const int k = (int)(i / Np);
    Wt[i] = (n < N && k < K) ? f2b(W[(long)n * K + k]) : f2b(0.f);
}

// sum-pool subwords into word slots: merged[b][w][e]; also emit time-major bf16 x0
__global__ void pool_sn_kernel(const float* __restrict__ bert, const int* __restrict__ wid,
                               float* __restrict__ merged, bf16_t* __restrict__ x0) {
    long i = (long)blockIdx.x * blockDim.x + threadIdx.x;
    const long total = (long)BATCH * SN * EMB;
    if (i >= total) return;
    const int e = (int)(i % EMB);
    const long r = i / EMB;
    const int w = (int)(r % SN);
    const int b = (int)(r / SN);
    float acc = 0.f;
    for (int s = 0; s < 96; ++s)
        if (wid[b * 96 + s] == w) acc += bert[((long)b * 96 + s) * EMB + e];
    merged[i] = acc;
    x0[((long)w * BATCH + b) * EMB + e] = f2b(acc);
}

__global__ void mask_kernel(const float* __restrict__ merged, int* __restrict__ mask) {
    int i = blockIdx.x * blockDim.x + threadIdx.x;
    if (i >= BATCH * SN) return;
    const float* m = merged + (long)i * EMB;
    float s = 0.f;
    for (int e = 0; e < EMB; ++e) s += m[e];
    mask[i] = (s != 0.f) ? 1 : 0;
}

__global__ void pool_sp_kernel(const float* __restrict__ wordEmb, const float* __restrict__ posEmb,
                               const int* __restrict__ widSp, const int* __restrict__ spEmd,
                               const int* __restrict__ spPos, float* __restrict__ d0) {
    long i = (long)blockIdx.x * blockDim.x + threadIdx.x;
    const long total = (long)BATCH * TDEC * EMB;
    if (i >= total) return;
    const int e = (int)(i % EMB);
    const long r = i / EMB;
    const int w = (int)(r % TDEC);
    const int b = (int)(r / TDEC);
    float acc = posEmb[(long)spPos[b * SPL + w] * EMB + e];
    for (int s = 0; s < TDEC; ++s)
        if (widSp[b * SPL + s] == w) acc += wordEmb[(long)spEmd[b * SPL + s] * EMB + e];
    d0[i] = acc;
}

// LayerNorm(768) + concat [landing, fix_dur, sub_emb(32)] -> dIn[t][b][DINP] bf16
__global__ __launch_bounds__(32) void ln_concat_kernel(
    const float* __restrict__ d0, const float* __restrict__ lnG, const float* __restrict__ lnB,
    const float* __restrict__ land, const float* __restrict__ dur,
    const int* __restrict__ subId, const float* __restrict__ subEmb, bf16_t* __restrict__ dIn) {
    const int row = blockIdx.x;          // b*39 + w
    const int b = row / TDEC, w = row % TDEC;
    const int lane = threadIdx.x;
    const float* x = d0 + (long)row * EMB;
    float s = 0.f;
    for (int e = lane; e < EMB; e += 32) s += x[e];
    for (int off = 16; off; off >>= 1) s += __shfl_xor(s, off, 32);
    const float mu = s / (float)EMB;
    float v = 0.f;
    for (int e = lane; e < EMB; e += 32) { float t = x[e] - mu; v += t * t; }
    for (int off = 16; off; off >>= 1) v += __shfl_xor(v, off, 32);
    const float inv = rsqrtf(v / (float)EMB + 1e-12f);
    bf16_t* drow = dIn + ((long)w * BATCH + b) * DINP;
    for (int e = lane; e < EMB; e += 32)
        drow[e] = f2b((x[e] - mu) * inv * lnG[e] + lnB[e]);
    if (lane == 0) {
        drow[768] = f2b(land[b * SPL + w]);
        drow[769] = f2b(dur[b * SPL + w]);
    }
    const int sid = subId[b];
    drow[770 + lane] = f2b(subEmb[sid * 32 + lane]);
}

// enc_out[b][s][0..127] = final x (bf16, time-major); [128] = sn_word_len
__global__ void encout_kernel(const bf16_t* __restrict__ x, const float* __restrict__ wlen,
                              float* __restrict__ encOut) {
    long i = (long)blockIdx.x * blockDim.x + threadIdx.x;
    const long total = (long)BATCH * SN * 129;
    if (i >= total) return;
    const int d = (int)(i % 129);
    const long r = i / 129;
    const int s = (int)(r % SN);
    const int b = (int)(r / SN);
    encOut[i] = (d < 128) ? b2f(x[((long)s * BATCH + b) * HID + d]) : wlen[b * SN + s];
}

// torch LSTMCell pointwise (gate order i,f,g,o); writes bf16 h (+optional copy)
__global__ void lstm_elem_kernel(const float* __restrict__ g, int ldg,
                                 float* __restrict__ c, bf16_t* __restrict__ h, int dh,
                                 bf16_t* __restrict__ extra, int ldExtra) {
    const int i = blockIdx.x * blockDim.x + threadIdx.x;
    if (i >= BATCH * dh) return;
    const int b = i / dh, k = i % dh;
    const float* gr = g + (long)b * ldg;
    const float iv = sigm(gr[k]);
    const float fv = sigm(gr[dh + k]);
    const float gv = tanhf(gr[2 * dh + k]);
    const float ov = sigm(gr[3 * dh + k]);
    const float cn = fv * c[i] + iv * gv;
    c[i] = cn;
    const float hv = ov * tanhf(cn);
    h[(long)b * dh + k] = f2b(hv);
    if (extra) extra[(long)b * ldExtra + k] = f2b(hv);
}

// windowed + gaussian attention for one decode step; one wave per batch row
__global__ __launch_bounds__(32) void attn_kernel(
    const float* __restrict__ q, const float* __restrict__ encOut,
    const int* __restrict__ mask, const int* __restrict__ spPos, int t,
    const bf16_t* __restrict__ h7, bf16_t* __restrict__ de, float* __restrict__ out) {
    const int b = blockIdx.x;
    const int lane = threadIdx.x;
    const int pos = spPos[b * SPL + t];
    float score = -3e38f;
    const bool valid = (lane < SN);
    if (valid) {
        const float* e = encOut + ((long)b * SN + lane) * 129;
        const float* qb = q + (long)b * 129;
        float acc = 0.f;
        for (int d = 0; d < 129; ++d) acc += qb[d] * e[d];
        const int left = (pos - 1 > 0) ? pos - 1 : 0;
        const int right = (pos + 1 < SN - 1) ? pos + 1 : SN - 1;
        const bool inval = (!mask[b * SN + lane]) || lane < left || lane > right;
        score = acc + (inval ? -1e9f : 0.f);
    }
    float m = score;
    for (int off = 16; off; off >>= 1) m = fmaxf(m, __shfl_xor(m, off, 32));
    float ex = valid ? __expf(score - m) : 0.f;
    float sum = ex;
    for (int off = 16; off; off >>= 1) sum += __shfl_xor(sum, off, 32);
    float w = 0.f;
    if (valid) {
        const float dd = (float)(lane - pos);
        w = (ex / sum) * __expf(-(dd * dd) / 0.5f);
        out[((long)b * TDEC + t) * OUTC + 45 + lane] = w;
    }
    for (int d = lane; d < 129; d += 32) {
        float acc = 0.f;
        for (int s2 = 0; s2 < SN; ++s2) {
            const float ws = __shfl(w, s2, 32);
            acc += ws * encOut[((long)b * SN + s2) * 129 + d];
        }
        de[(long)b * DEP + 45 + d] = f2b(acc);     // ctx -> de[45..173]
    }
    for (int d = lane; d < 128; d += 32)
        de[(long)b * DEP + 174 + d] = h7[(long)b * HID + d];  // h7 -> de[174..301]
}

// ---------------------------------------------------------------------------
// Host side
// ---------------------------------------------------------------------------
struct PW { bf16_t* p; int np; int kp; };

static void gemm(hipStream_t st, const bf16_t* A, int lda, PW w, const float* bias,
                 int M, int N, int flags, float* oF, int ldf, bf16_t* oB, int ldb) {
    dim3 grid((N + 127) / 128, M / 64);
    gemm_bf16_wmma_kernel<<<grid, 256, 0, st>>>(A, lda, w.p, w.np, bias,
                                                M, N, w.kp, flags, oF, ldf, oB, ldb);
}
static void launch_zero(hipStream_t st, void* p, size_t bytes) {
    long n = (long)(bytes / 4);
    zero_u32_kernel<<<(unsigned)((n + 255) / 256), 256, 0, st>>>((unsigned int*)p, n);
}

// d_in layout (setup_inputs dict order; params flattened in insertion order):
//  0 sn_bert_out 1 sn_word_len 2 sp_fix_dur 3 sp_landing_pos 4 word_ids_sn
//  5 sp_emd 6 sp_pos 7 word_ids_sp 8 sub_id
//  9 word_emb 10 pos_emb 11 ln_g 12 ln_b 13 sub_emb
//  14..77  enc_lstm: layer l { fwd:{wih,whh,bih,bhh}, bwd:{...} }
//  78..109 dec_cells: cell j {wih,whh,bih,bhh}
//  110 attn.w 111 attn.b
//  112..121 dense (w,b)x5   122..131 duration (w,b)x5   132..139 landing (w,b)x4
extern "C" void kernel_launch(void* const* d_in, const int* in_sizes, int n_in,
                              void* d_out, int out_size, void* d_ws, size_t ws_size,
                              hipStream_t stream) {
    auto F = [&](int i) { return (const float*)d_in[i]; };
    auto I = [&](int i) { return (const int*)d_in[i]; };
    char* base = (char*)d_ws;
    size_t off = 0;
    auto alloc = [&](size_t bytes) -> char* {
        char* p = base + off;
        off = (off + bytes + 255) & ~(size_t)255;
        return p;
    };

    // ---- activations / scratch ----
    float*  d0f   = (float*)alloc((size_t)BATCH * TDEC * EMB * 4);  // merged_sn then d0
    int*    maskB = (int*)alloc((size_t)BATCH * SN * 4);
    bf16_t* x0    = (bf16_t*)alloc((size_t)SN * BATCH * EMB * 2);
    bf16_t* xA    = (bf16_t*)alloc((size_t)SN * BATCH * HID * 2);
    bf16_t* xB    = (bf16_t*)alloc((size_t)SN * BATCH * HID * 2);
    float*  Xg    = (float*)alloc((size_t)SN * BATCH * 256 * 4);    // also decoder gate tmp
    bf16_t* hsE   = (bf16_t*)alloc((size_t)BATCH * H2 * 2);
    float*  csE   = (float*)alloc((size_t)BATCH * H2 * 4);
    float*  encOut= (float*)alloc((size_t)BATCH * SN * 129 * 4);
    bf16_t* dIn   = (bf16_t*)alloc((size_t)TDEC * BATCH * DINP * 2);
    float*  Xg0   = (float*)alloc((size_t)TDEC * BATCH * 512 * 4);
    bf16_t* hD    = (bf16_t*)alloc((size_t)8 * BATCH * HID * 2);
    float*  cD    = (float*)alloc((size_t)8 * BATCH * HID * 4);
    float*  qbuf  = (float*)alloc((size_t)BATCH * 129 * 4);
    bf16_t* de    = (bf16_t*)alloc((size_t)BATCH * DEP * 2);
    bf16_t* m1    = (bf16_t*)alloc((size_t)BATCH * 512 * 2);
    bf16_t* m2    = (bf16_t*)alloc((size_t)BATCH * 512 * 2);

    // ---- weight pack-transpose (f32 [N][K] -> bf16 W^T [Kp][Np], L2-resident) ----
    auto pack = [&](const float* W, int N, int K) -> PW {
        const int Kp = (K + 31) & ~31;
        const int Np = (N + 127) & ~127;
        bf16_t* d = (bf16_t*)alloc((size_t)Kp * Np * 2);
        long n = (long)Kp * Np;
        pack_wt_kernel<<<(unsigned)((n + 255) / 256), 256, 0, stream>>>(W, d, N, K, Kp, Np);
        return {d, Np, Kp};
    };
    auto biasSum = [&](const float* a, const float* b, int n) -> float* {
        float* o = (float*)alloc((size_t)n * 4);
        add_f32_kernel<<<(n + 255) / 256, 256, 0, stream>>>(a, b, o, n);
        return o;
    };

    PW encWih[8][2], encWhh[8][2];
    float* encB[8][2];
    for (int l = 0; l < 8; ++l)
        for (int d = 0; d < 2; ++d) {
            const int bi = 14 + 8 * l + 4 * d;
            const int din = l ? HID : EMB;
            encWih[l][d] = pack(F(bi + 0), 256, din);
            encWhh[l][d] = pack(F(bi + 1), 256, H2);
            encB[l][d] = biasSum(F(bi + 2), F(bi + 3), 256);
        }
    PW decWih[8], decWhh[8];
    float* decB[8];
    for (int j = 0; j < 8; ++j) {
        const int bi = 78 + 4 * j;
        const int din = j ? HID : 802;
        decWih[j] = pack(F(bi + 0), 512, din);
        decWhh[j] = pack(F(bi + 1), 512, HID);
        decB[j] = biasSum(F(bi + 2), F(bi + 3), 512);
    }
    PW attnW = pack(F(110), 129, HID);
    const float* attnB = F(111);

    const int denseD[5][2] = {{512, 257}, {256, 512}, {256, 256}, {256, 256}, {45, 256}};
    const int durD[5][2]   = {{512, 302}, {256, 512}, {256, 256}, {256, 256}, {1, 256}};
    const int landD[4][2]  = {{256, 302}, {128, 256}, {128, 128}, {1, 128}};
    PW denseW[5]; const float* denseBs[5];
    PW durW[5];   const float* durBs[5];
    PW landW[4];  const float* landBs[4];
    for (int i = 0; i < 5; ++i) { denseW[i] = pack(F(112 + 2 * i), denseD[i][0], denseD[i][1]); denseBs[i] = F(113 + 2 * i); }
    for (int i = 0; i < 5; ++i) { durW[i]   = pack(F(122 + 2 * i), durD[i][0], durD[i][1]);     durBs[i]   = F(123 + 2 * i); }
    for (int i = 0; i < 4; ++i) { landW[i]  = pack(F(132 + 2 * i), landD[i][0], landD[i][1]);   landBs[i]  = F(133 + 2 * i); }

    // zero padded activation buffers once per call (pad columns must stay 0)
    launch_zero(stream, dIn, (size_t)TDEC * BATCH * DINP * 2);
    launch_zero(stream, de, (size_t)BATCH * DEP * 2);

    // ---- encoder input pooling + mask ----
    {
        long n = (long)BATCH * SN * EMB;
        pool_sn_kernel<<<(unsigned)((n + 255) / 256), 256, 0, stream>>>(F(0), I(4), d0f, x0);
        mask_kernel<<<(BATCH * SN + 255) / 256, 256, 0, stream>>>(d0f, maskB);
    }

    // ---- 8-layer BiLSTM encoder ----
    const bf16_t* xin = x0;
    bf16_t* xout = xA;
    int din = EMB;
    for (int l = 0; l < 8; ++l) {
        for (int d = 0; d < 2; ++d) {
            // hoisted input projection for all 24 steps at once (M = 24*512)
            gemm(stream, xin, din, encWih[l][d], encB[l][d],
                 SN * BATCH, 256, 0, Xg, 256, nullptr, 0);
            launch_zero(stream, hsE, (size_t)BATCH * H2 * 2);
            launch_zero(stream, csE, (size_t)BATCH * H2 * 4);
            for (int i = 0; i < SN; ++i) {
                const int tr = d ? (SN - 1 - i) : i;
                float* g = Xg + (size_t)tr * BATCH * 256;
                gemm(stream, hsE, H2, encWhh[l][d], nullptr,
                     BATCH, 256, 2, g, 256, nullptr, 0);  // accumulate h*Whh^T
                lstm_elem_kernel<<<(BATCH * H2 + 255) / 256, 256, 0, stream>>>(
                    g, 256, csE, hsE, H2, xout + (size_t)tr * BATCH * HID + d * H2, HID);
            }
        }
        xin = xout;
        xout = (xout == xA) ? xB : xA;
        din = HID;
    }

    // ---- enc_out = [h(128) | sn_word_len] in f32 ----
    {
        long n = (long)BATCH * SN * 129;
        encout_kernel<<<(unsigned)((n + 255) / 256), 256, 0, stream>>>((const bf16_t*)xin, F(1), encOut);
    }

    // ---- decoder input embeddings ----
    {
        long n = (long)BATCH * TDEC * EMB;
        pool_sp_kernel<<<(unsigned)((n + 255) / 256), 256, 0, stream>>>(F(9), F(10), I(7), I(5), I(6), d0f);
        ln_concat_kernel<<<BATCH * TDEC, 32, 0, stream>>>(d0f, F(11), F(12), F(3), F(2), I(8), F(13), dIn);
    }

    // cell-0 input projection hoisted over all 39 steps (M = 39*512, K = 802->832)
    gemm(stream, dIn, DINP, decWih[0], decB[0], TDEC * BATCH, 512, 0, Xg0, 512, nullptr, 0);
    launch_zero(stream, hD, (size_t)8 * BATCH * HID * 2);
    launch_zero(stream, cD, (size_t)8 * BATCH * HID * 4);

    float* outF = (float*)d_out;
    for (int t = 0; t < TDEC; ++t) {
        // cell 0: accumulate recurrent term onto precomputed gates
        float* g0 = Xg0 + (size_t)t * BATCH * 512;
        gemm(stream, hD, HID, decWhh[0], nullptr, BATCH, 512, 2, g0, 512, nullptr, 0);
        lstm_elem_kernel<<<(BATCH * HID + 255) / 256, 256, 0, stream>>>(g0, 512, cD, hD, HID, nullptr, 0);
        // cells 1..7
        for (int j = 1; j < 8; ++j) {
            bf16_t* hin = hD + (size_t)(j - 1) * BATCH * HID;
            bf16_t* hj  = hD + (size_t)j * BATCH * HID;
            float*  cj  = cD + (size_t)j * BATCH * HID;
            gemm(stream, hin, HID, decWih[j], decB[j], BATCH, 512, 0, Xg, 512, nullptr, 0);
            gemm(stream, hj,  HID, decWhh[j], nullptr, BATCH, 512, 2, Xg, 512, nullptr, 0);
            lstm_elem_kernel<<<(BATCH * HID + 255) / 256, 256, 0, stream>>>(Xg, 512, cj, hj, HID, nullptr, 0);
        }
        bf16_t* h7 = hD + (size_t)7 * BATCH * HID;
        // attention query + masked gaussian attention; fills de[45..301]
        gemm(stream, h7, HID, attnW, attnB, BATCH, 129, 0, qbuf, 129, nullptr, 0);
        attn_kernel<<<BATCH, 32, 0, stream>>>(qbuf, encOut, maskB, I(6), t, h7, de, outF);
        // dense head (loc) -> out[.,t,0..44] and de[0..44]
        gemm(stream, de + 45, DEP, denseW[0], denseBs[0], BATCH, 512, 1, nullptr, 0, m1, 512);
        gemm(stream, m1, 512, denseW[1], denseBs[1], BATCH, 256, 1, nullptr, 0, m2, 256);
        gemm(stream, m2, 256, denseW[2], denseBs[2], BATCH, 256, 1, nullptr, 0, m1, 256);
        gemm(stream, m1, 256, denseW[3], denseBs[3], BATCH, 256, 1, nullptr, 0, m2, 256);
        gemm(stream, m2, 256, denseW[4], denseBs[4], BATCH, 45, 0,
             outF + t * OUTC, TDEC * OUTC, de, DEP);
        // duration head -> out[.,t,69]
        gemm(stream, de, DEP, durW[0], durBs[0], BATCH, 512, 1, nullptr, 0, m1, 512);
        gemm(stream, m1, 512, durW[1], durBs[1], BATCH, 256, 1, nullptr, 0, m2, 256);
        gemm(stream, m2, 256, durW[2], durBs[2], BATCH, 256, 1, nullptr, 0, m1, 256);
        gemm(stream, m1, 256, durW[3], durBs[3], BATCH, 256, 1, nullptr, 0, m2, 256);
        gemm(stream, m2, 256, durW[4], durBs[4], BATCH, 1, 0,
             outF + t * OUTC + 69, TDEC * OUTC, nullptr, 0);
        // landing head -> out[.,t,70]
        gemm(stream, de, DEP, landW[0], landBs[0], BATCH, 256, 1, nullptr, 0, m1, 256);
        gemm(stream, m1, 256, landW[1], landBs[1], BATCH, 128, 1, nullptr, 0, m2, 128);
        gemm(stream, m2, 128, landW[2], landBs[2], BATCH, 128, 1, nullptr, 0, m1, 128);
        gemm(stream, m1, 128, landW[3], landBs[3], BATCH, 1, 0,
             outF + t * OUTC + 70, TDEC * OUTC, nullptr, 0);
    }
    (void)in_sizes; (void)n_in; (void)out_size; (void)ws_size;
}
